// MyRGCNConv_35536559407731
// MI455X (gfx1250) — matrix operand order repaired and checked by reading, not compiled, not run on hardware
//
#include <hip/hip_runtime.h>

// RGCN forward for MI455X (gfx1250, wave32, WMMA).
// Pre-pass: convert X -> bf16 hi/lo (row-major) and W -> bf16 hi/lo transposed [n][k].
// Per relation r: T = X @ W[r] via split-bf16 WMMA GEMM (fp32-quality:
// Xh*Wh + Xh*Wl + Xl*Wh), then gather/segment-sum edges of relation r into out;
// divide by deg on the last pass.

typedef __attribute__((ext_vector_type(16))) __bf16 v16bf;
typedef __attribute__((ext_vector_type(8)))  float  v8f;

#define DIM   256     // D_IN == D_HID == 256
#define BM    64
#define BN    128
#define BK    32
#define BKP   40      // padded LDS row: 80B -> 16B-aligned fragments, conflict-free

union FragU { uint4 u4[2]; v16bf v; };
union Pack4 { __bf16 b[4]; uint2 u; };

// ---------------------------------------------------------------------------
// One-shot: X (f32) -> Xh/Xl (bf16), same row-major layout.
// ---------------------------------------------------------------------------
__global__ __launch_bounds__(256)
void convert_x(const float* __restrict__ X,
               __bf16* __restrict__ Xh, __bf16* __restrict__ Xl, int total4)
{
    int i = blockIdx.x * 256 + threadIdx.x;
    if (i >= total4) return;
    float4 f = ((const float4*)X)[i];
    float vals[4] = {f.x, f.y, f.z, f.w};
    Pack4 h, l;
    #pragma unroll
    for (int j = 0; j < 4; ++j) {
        h.b[j] = (__bf16)vals[j];
        l.b[j] = (__bf16)(vals[j] - (float)h.b[j]);
    }
    ((uint2*)Xh)[i] = h.u;
    ((uint2*)Xl)[i] = l.u;
}

// ---------------------------------------------------------------------------
// One-shot: W[r][k][n] (f32) -> Wh/Wl[r][n][k] (bf16, transposed per relation).
// One thread per (r, k-quad, n); lanes sweep n for coalesced reads.
// ---------------------------------------------------------------------------
__global__ __launch_bounds__(256)
void convert_wt(const float* __restrict__ W,
                __bf16* __restrict__ Wh, __bf16* __restrict__ Wl, int total)
{
    int id = blockIdx.x * 256 + threadIdx.x;   // [0, R*64*256)
    if (id >= total) return;
    int n   = id & 255;
    int kqr = id >> 8;
    int kq  = kqr & 63;                        // k/4
    int r   = kqr >> 6;
    Pack4 h, l;
    #pragma unroll
    for (int j = 0; j < 4; ++j) {
        float v = W[(size_t)r * DIM * DIM + (size_t)(kq * 4 + j) * DIM + n];
        h.b[j] = (__bf16)v;
        l.b[j] = (__bf16)(v - (float)h.b[j]);
    }
    size_t o = ((size_t)r * DIM + n) * DIM + kq * 4;   // [r][n][k]
    *(uint2*)(Wh + o) = h.u;
    *(uint2*)(Wl + o) = l.u;
}

// ---------------------------------------------------------------------------
// GEMM: T[M,256] = X[M,256] @ W_r[256,256] with pre-split bf16 operands.
// Block: 128 threads = 4 waves; tile 64x128, K-step 32; 24 WMMAs/wave/K-step.
// All LDS traffic is b128.
// ---------------------------------------------------------------------------
__global__ __launch_bounds__(128)
void rgcn_gemm_bf16split(const __bf16* __restrict__ Xh,
                         const __bf16* __restrict__ Xl,
                         const __bf16* __restrict__ Wh,   // [n][k] for this r
                         const __bf16* __restrict__ Wl,
                         float* __restrict__ T,
                         int M)
{
    __shared__ __align__(16) __bf16 AsH[BM][BKP];   // [m][k]
    __shared__ __align__(16) __bf16 AsL[BM][BKP];
    __shared__ __align__(16) __bf16 BsH[BN][BKP];   // [n][k]
    __shared__ __align__(16) __bf16 BsL[BN][BKP];

    const int tid  = threadIdx.x;
    const int lane = tid & 31;
    const int wave = tid >> 5;                 // 0..3
    const int m0   = blockIdx.x * BM;
    const int n0   = blockIdx.y * BN;

    v8f acc[BN / 16] = {};                     // 8 x (16x16 f32)

    // ISA 7.12.2 fragment coordinates (wave32)
    const int arow = wave * 16 + (lane & 15);  // A: M row
    const int akb  = (lane >> 4) * 8;          // A: K base (elems 0-7 -> akb.., 8-15 -> akb+16..)
    const int bcol = lane & 15;                // B: N column
    const int bkb  = (lane >> 4) * 8;          // B: K base (elems 0-7 -> bkb.., 8-15 -> bkb+16..)

    // A staging mapping: 2 threads per row, 32B each half
    const int asrow = tid >> 1;
    const int ashlf = (tid & 1) * 16;          // k offset of this half

    for (int k0 = 0; k0 < DIM; k0 += BK) {
        // ---- stage A: 64x32 bf16 hi/lo, b128 traffic ----
        {
            int gr = m0 + asrow;
            uint4 h0 = {0,0,0,0}, h1 = {0,0,0,0}, l0 = {0,0,0,0}, l1 = {0,0,0,0};
            if (gr < M) {
                const uint4* ph = (const uint4*)(Xh + (size_t)gr * DIM + k0 + ashlf);
                const uint4* pl = (const uint4*)(Xl + (size_t)gr * DIM + k0 + ashlf);
                h0 = ph[0]; h1 = ph[1];
                l0 = pl[0]; l1 = pl[1];
            }
            *(uint4*)&AsH[asrow][ashlf]     = h0;
            *(uint4*)&AsH[asrow][ashlf + 8] = h1;
            *(uint4*)&AsL[asrow][ashlf]     = l0;
            *(uint4*)&AsL[asrow][ashlf + 8] = l1;
        }
        // ---- stage B: 128x32 bf16 hi/lo, one n-row (64B) per thread ----
        {
            const uint4* ph = (const uint4*)(Wh + (size_t)(n0 + tid) * DIM + k0);
            const uint4* pl = (const uint4*)(Wl + (size_t)(n0 + tid) * DIM + k0);
            #pragma unroll
            for (int j = 0; j < 4; ++j) {
                *(uint4*)&BsH[tid][j * 8] = ph[j];
                *(uint4*)&BsL[tid][j * 8] = pl[j];
            }
        }
        __syncthreads();

        // ---- A fragments: 2 x b128 each ----
        FragU aH, aL;
        aH.u4[0] = *(const uint4*)&AsH[arow][akb];
        aH.u4[1] = *(const uint4*)&AsH[arow][akb + 16];
        aL.u4[0] = *(const uint4*)&AsL[arow][akb];
        aL.u4[1] = *(const uint4*)&AsL[arow][akb + 16];

        // ---- 8 N-tiles x 3 WMMAs ----
        #pragma unroll
        for (int t = 0; t < BN / 16; ++t) {
            int col = t * 16 + bcol;
            FragU bH, bL;
            bH.u4[0] = *(const uint4*)&BsH[col][bkb];
            bH.u4[1] = *(const uint4*)&BsH[col][bkb + 16];
            bL.u4[0] = *(const uint4*)&BsL[col][bkb];
            bL.u4[1] = *(const uint4*)&BsL[col][bkb + 16];
            acc[t] = __builtin_amdgcn_wmma_f32_16x16x32_bf16(
                false, aH.v, false, bH.v, (short)0, acc[t], false, false);
            acc[t] = __builtin_amdgcn_wmma_f32_16x16x32_bf16(
                false, aH.v, false, bL.v, (short)0, acc[t], false, false);
            acc[t] = __builtin_amdgcn_wmma_f32_16x16x32_bf16(
                false, aL.v, false, bH.v, (short)0, acc[t], false, false);
        }
        __syncthreads();
    }

    // ---- store C/D: VGPR v -> M = 8*lane[4] + v, N = lane&15 ----
    const int nl = lane & 15;
    const int mb = (lane >> 4) * 8;
    #pragma unroll
    for (int t = 0; t < BN / 16; ++t) {
        int gc = n0 + t * 16 + nl;
        #pragma unroll
        for (int v = 0; v < 8; ++v) {
            int gr = m0 + wave * 16 + mb + v;
            if (gr < M) T[(size_t)gr * DIM + gc] = acc[t][v];
        }
    }
}

// ---------------------------------------------------------------------------
// Aggregation: one block per center, one channel per thread. Deterministic
// (uniform scalar edge loop, no atomics). Last pass scales by 1/deg.
// ---------------------------------------------------------------------------
__global__ __launch_bounds__(256)
void rgcn_agg(const float* __restrict__ T,
              const int* __restrict__ ptr,
              const int* __restrict__ idx,
              const int* __restrict__ et,
              float* __restrict__ out,
              int rel, int first, int last)
{
    const int c = blockIdx.x;
    const int h = threadIdx.x;                // channel
    const int e0  = ptr[c];
    const int e1  = ptr[c + 1];
    const int deg = e1 - e0;

    float acc = 0.f;
    for (int e = e0; e < e1; ++e) {           // uniform across block -> scalarizes
        if (et[e] == rel)
            acc += T[(size_t)idx[e] * DIM + h];
    }

    float r = first ? acc : (out[(size_t)c * DIM + h] + acc);
    if (last && deg > 0) r *= (1.0f / (float)deg);
    out[(size_t)c * DIM + h] = r;
}

// ---------------------------------------------------------------------------
// Launch. Workspace layout (bytes):
//   T   : M*256*4                     (51.2 MB)
//   Xh  : M*256*2, Xl : M*256*2       (51.2 MB)
//   Wh  : R*256*256*2, Wl : same      ( 2.0 MB)
// ---------------------------------------------------------------------------
extern "C" void kernel_launch(void* const* d_in, const int* in_sizes, int n_in,
                              void* d_out, int out_size, void* d_ws, size_t ws_size,
                              hipStream_t stream)
{
    const float* x       = (const float*)d_in[0];   // [M, 256] f32
    const float* weights = (const float*)d_in[1];   // [R, 256, 256] f32
    const int*   ptr     = (const int*)  d_in[2];   // [C+1]
    const int*   idx     = (const int*)  d_in[3];   // [E]
    const int*   et      = (const int*)  d_in[4];   // [E]

    const int M       = in_sizes[0] / DIM;
    const int R       = in_sizes[1] / (DIM * DIM);
    const int nCenter = in_sizes[2] - 1;

    char* ws = (char*)d_ws;
    float*  T  = (float*) ws;                       ws += (size_t)M * DIM * 4;
    __bf16* Xh = (__bf16*)ws;                       ws += (size_t)M * DIM * 2;
    __bf16* Xl = (__bf16*)ws;                       ws += (size_t)M * DIM * 2;
    __bf16* Wh = (__bf16*)ws;                       ws += (size_t)R * DIM * DIM * 2;
    __bf16* Wl = (__bf16*)ws;
    float* out = (float*)d_out;

    // one-shot conversions
    {
        int total4 = M * DIM / 4;
        convert_x<<<(total4 + 255) / 256, 256, 0, stream>>>(x, Xh, Xl, total4);
        int totalw = R * (DIM / 4) * DIM;
        convert_wt<<<(totalw + 255) / 256, 256, 0, stream>>>(weights, Wh, Wl, totalw);
    }

    dim3 ggrid((M + BM - 1) / BM, DIM / BN);        // (782, 2)
    for (int r = 0; r < R; ++r) {
        rgcn_gemm_bf16split<<<ggrid, 128, 0, stream>>>(
            Xh, Xl, Wh + (size_t)r * DIM * DIM, Wl + (size_t)r * DIM * DIM, T, M);
        rgcn_agg<<<nCenter, 256, 0, stream>>>(
            T, ptr, idx, et, out, r, (r == 0) ? 1 : 0, (r == R - 1) ? 1 : 0);
    }
}
